// CustomCrossAttention_84645215470343
// MI455X (gfx1250) — matrix operand order repaired and checked by reading, not compiled, not run on hardware
//
#include <hip/hip_runtime.h>
#include <hip/hip_bf16.h>

// ---------------- WMMA types (gfx1250, wave32) ----------------
typedef __attribute__((ext_vector_type(16))) __bf16 v16bf;
typedef __attribute__((ext_vector_type(8)))  float  v8f;

union FragU { uint4 u4[2]; v16bf v; };

enum { EPI_BF16 = 0, EPI_BF16_T = 1, EPI_F32 = 2, EPI_GATED = 3, EPI_MAHA = 4, EPI_ACC = 5 };

struct GemmP {
  const __bf16* A;  unsigned long long sA; int lda;
  const __bf16* Bm; unsigned long long sB; int ldb;
  const float* bias;
  __hip_bfloat16* outB; float* outF; unsigned long long sO; int ldo;
  int M, N, K;
  float alpha, diagAdd;
  const float* colScale; unsigned long long sCol;
  const float* gate; unsigned long long sGate;
  const float* skip; unsigned long long sSkip; int ldskip;
  float* maha; unsigned long long sMaha;
  int mode;
};

__device__ __forceinline__ void epi_store(const GemmP& p, int b, const v8f& acc, int m0, int mhi, int n) {
#pragma unroll
  for (int r = 0; r < 8; ++r) {
    int m = m0 + mhi + r;
    float v = acc[r];
    if (p.mode == EPI_BF16) {
      if (p.bias) v += p.bias[n];
      p.outB[(unsigned long long)b * p.sO + (unsigned long long)m * p.ldo + n] = __float2bfloat16(v);
    } else if (p.mode == EPI_BF16_T) {
      if (p.bias) v += p.bias[n];
      p.outB[(unsigned long long)b * p.sO + (unsigned long long)n * p.ldo + m] = __float2bfloat16(v);
    } else if (p.mode == EPI_F32) {
      v *= p.alpha;
      if (p.colScale) v *= p.colScale[(unsigned long long)b * p.sCol + n];
      if (m == n) v += p.diagAdd;
      p.outF[(unsigned long long)b * p.sO + (unsigned long long)m * p.ldo + n] = v;
    } else if (p.mode == EPI_GATED) {
      float g   = p.gate[(unsigned long long)b * p.sGate + m];
      float lin = v + (p.bias ? p.bias[n] : 0.f);
      float sk  = p.skip[(unsigned long long)b * p.sSkip + (unsigned long long)m * p.ldskip + n];
      p.outF[(unsigned long long)b * p.sO + (unsigned long long)m * p.ldo + n] = g * sk + (1.f - g) * lin;
    } else if (p.mode == EPI_MAHA) {
      atomicAdd(&p.maha[(unsigned long long)b * p.sMaha + m], v * v);
    } else { // EPI_ACC : out += acc
      float* o = &p.outF[(unsigned long long)b * p.sO + (unsigned long long)m * p.ldo + n];
      *o = *o + v;
    }
  }
}

// NT GEMM: D[m][n] = sum_k A[m][k] * B[n][k]  (bf16 in, fp32 accumulate)
// Requires M%64==0, N%64==0, K%32==0.  8 waves/WG -> 64x64 macro tile, each
// wave computes 16x32 with 2 v_wmma_f32_16x16x32_bf16 per K-step (A reuse).
__global__ __launch_bounds__(256) void gemm_bf16_wmma(GemmP p) {
  int b    = blockIdx.z;
  int lane = threadIdx.x & 31;
  int wave = threadIdx.x >> 5;
  int wm = wave & 3, wn = wave >> 2;
  int m0 = blockIdx.x * 64 + wm * 16;
  int n0 = blockIdx.y * 64 + wn * 32;
  const __bf16* Ab = p.A  + (unsigned long long)b * p.sA;
  const __bf16* Bb = p.Bm + (unsigned long long)b * p.sB;
  int mrow  = m0 + (lane & 15);
  int nrow0 = n0 + (lane & 15);
  int khalfA = (lane & 16) ? 8 : 0;    // A: K split {khalf..+7, 16+khalf..+7}
  int khalfB = (lane & 16) ? 8 : 0;    // B: K split {khalf..+7, 16+khalf..+7}
  const __bf16* pa  = Ab + (unsigned long long)mrow  * p.lda + khalfA;
  const __bf16* pb0 = Bb + (unsigned long long)nrow0 * p.ldb + khalfB;
  const __bf16* pb1 = pb0 + (unsigned long long)16 * p.ldb;
  v8f acc0 = {}; v8f acc1 = {};
  for (int k = 0; k < p.K; k += 32) {
    FragU fa, fb0, fb1;
    fa.u4[0]  = *(const uint4*)(pa  + k);
    fa.u4[1]  = *(const uint4*)(pa  + k + 16);
    fb0.u4[0] = *(const uint4*)(pb0 + k);
    fb0.u4[1] = *(const uint4*)(pb0 + k + 16);
    fb1.u4[0] = *(const uint4*)(pb1 + k);
    fb1.u4[1] = *(const uint4*)(pb1 + k + 16);
    acc0 = __builtin_amdgcn_wmma_f32_16x16x32_bf16(false, fa.v, false, fb0.v, (short)0, acc0, false, false);
    acc1 = __builtin_amdgcn_wmma_f32_16x16x32_bf16(false, fa.v, false, fb1.v, (short)0, acc1, false, false);
  }
  int mhi = (lane & 16) ? 8 : 0;
  epi_store(p, b, acc0, m0, mhi, nrow0);
  epi_store(p, b, acc1, m0, mhi, nrow0 + 16);
}

// ---------------- elementwise / reduction kernels ----------------
__global__ void f2b_kernel(const float* __restrict__ in, __hip_bfloat16* __restrict__ out, size_t n) {
  size_t i = (size_t)blockIdx.x * blockDim.x + threadIdx.x;
  size_t st = (size_t)gridDim.x * blockDim.x;
  for (; i < n; i += st) out[i] = __float2bfloat16(in[i]);
}

__global__ void zero_kernel(float* __restrict__ out, size_t n) {
  size_t i = (size_t)blockIdx.x * blockDim.x + threadIdx.x;
  size_t st = (size_t)gridDim.x * blockDim.x;
  for (; i < n; i += st) out[i] = 0.f;
}

__global__ void mean_kernel(const float* __restrict__ X, float* __restrict__ mu, int T, int D) {
  int b = blockIdx.y;
  int d = blockIdx.x * blockDim.x + threadIdx.x;
  if (d >= D) return;
  const float* p = X + (size_t)b * T * D + d;
  float s = 0.f;
  for (int t = 0; t < T; ++t) s += p[(size_t)t * D];
  mu[(size_t)b * D + d] = s / (float)T;
}

// centered bf16 (B,T,D) and transposed centered bf16 (B,D,T) via LDS tile
__global__ __launch_bounds__(1024) void center_tb_kernel(const float* __restrict__ X, const float* __restrict__ mu,
                                                         __hip_bfloat16* __restrict__ Xc,
                                                         __hip_bfloat16* __restrict__ XcT, int T, int D) {
  __shared__ float tile[32][33];
  int b = blockIdx.z;
  int d0 = blockIdx.x * 32, t0 = blockIdx.y * 32;
  int tx = threadIdx.x, ty = threadIdx.y;
  int d = d0 + tx, t = t0 + ty;
  float v = X[((size_t)b * T + t) * D + d] - mu[(size_t)b * D + d];
  Xc[((size_t)b * T + t) * D + d] = __float2bfloat16(v);
  tile[ty][tx] = v;
  __syncthreads();
  XcT[((size_t)b * D + (d0 + ty)) * T + (t0 + tx)] = __float2bfloat16(tile[tx][ty]);
}

// per-batch right-looking Cholesky in place (lower), then zero upper triangle
__global__ __launch_bounds__(256) void cholesky_kernel(float* __restrict__ S, unsigned long long sB, int d) {
  float* M = S + (unsigned long long)blockIdx.x * sB;
  int tid = threadIdx.x, nt = blockDim.x;
  for (int j = 0; j < d; ++j) {
    float djj = M[(size_t)j * d + j];
    __syncthreads();
    float l = sqrtf(fmaxf(djj, 1e-30f));
    float inv = 1.0f / l;
    if (tid == 0) M[(size_t)j * d + j] = l;
    for (int i = j + 1 + tid; i < d; i += nt) M[(size_t)i * d + j] *= inv;
    __syncthreads();
    for (int i = j + 1 + tid; i < d; i += nt) {
      float lij = M[(size_t)i * d + j];
      float* row = M + (size_t)i * d;
      for (int k = j + 1; k <= i; ++k) row[k] = fmaf(-lij, M[(size_t)k * d + j], row[k]);
    }
    __syncthreads();
  }
  for (int i = tid; i < d; i += nt)
    for (int k = i + 1; k < d; ++k) M[(size_t)i * d + k] = 0.f;
}

// Minv = L^-1 (lower), per-batch block, one column per thread (L2 resident)
__global__ __launch_bounds__(256) void trinv_kernel(const float* __restrict__ L, float* __restrict__ Minv,
                                                    unsigned long long sB, int d) {
  const float* Lb = L + (unsigned long long)blockIdx.x * sB;
  float* Mb = Minv + (unsigned long long)blockIdx.x * sB;
  for (int j = threadIdx.x; j < d; j += blockDim.x) {
    for (int i = 0; i < j; ++i) Mb[(size_t)i * d + j] = 0.f;
    for (int i = j; i < d; ++i) {
      float s = (i == j) ? 1.0f : 0.0f;
      const float* Lrow = Lb + (size_t)i * d;
      for (int k = j; k < i; ++k) s -= Lrow[k] * Mb[(size_t)k * d + j];
      Mb[(size_t)i * d + j] = s / Lrow[i];
    }
  }
}

__global__ __launch_bounds__(256) void logdet_kernel(const float* __restrict__ L, float* __restrict__ out, int d) {
  __shared__ float red[256];
  int b = blockIdx.x;
  float s = 0.f;
  for (int i = threadIdx.x; i < d; i += blockDim.x)
    s += logf(L[(size_t)b * d * d + (size_t)i * d + i]);
  red[threadIdx.x] = s;
  __syncthreads();
  for (int o = 128; o > 0; o >>= 1) { if (threadIdx.x < o) red[threadIdx.x] += red[threadIdx.x + o]; __syncthreads(); }
  if (threadIdx.x == 0) out[b] = 2.0f * red[0];
}

__global__ void pdf_kernel(const float* __restrict__ maha, const float* __restrict__ logdet,
                           float* __restrict__ pdf, int d, int T, int n) {
  int i = blockIdx.x * blockDim.x + threadIdx.x;
  if (i >= n) return;
  int b = i / T;
  float mh = fminf(maha[i], 100.0f);
  float lp = -0.5f * ((float)d * 1.8378770664093453f + logdet[b] + mh);
  lp = fminf(fmaxf(lp, -50.0f), 50.0f);
  float w = expf(lp);
  pdf[i] = fminf(fmaxf(w, 1e-10f), 1e10f);
}

// one wave per row; T <= 1024 (32 regs per lane).  Includes reference's /Z
__global__ __launch_bounds__(256) void softmax_kernel(const float* __restrict__ S, __hip_bfloat16* __restrict__ P,
                                                      int T, int rows) {
  int lane = threadIdx.x & 31;
  int row = blockIdx.x * 8 + (threadIdx.x >> 5);
  if (row >= rows) return;
  const float* s = S + (size_t)row * T;
  float v[32];
  int cnt = T >> 5;
  float mx = -3.0e38f;
  for (int i = 0; i < cnt; ++i) { v[i] = s[i * 32 + lane]; mx = fmaxf(mx, v[i]); }
  for (int o = 16; o > 0; o >>= 1) mx = fmaxf(mx, __shfl_xor(mx, o, 32));
  float sum = 0.f;
  for (int i = 0; i < cnt; ++i) { v[i] = expf(v[i] - mx); sum += v[i]; }
  for (int o = 16; o > 0; o >>= 1) sum += __shfl_xor(sum, o, 32);
  float inv = 1.0f / (sum * (1.0f + 1e-8f));
  __hip_bfloat16* o = P + (size_t)row * T;
  for (int i = 0; i < cnt; ++i) o[i * 32 + lane] = __float2bfloat16(v[i] * inv);
}

// g = sigmoid([X | R] . W + b) , one wave per token row
__global__ __launch_bounds__(256) void gate_kernel(const float* __restrict__ X, const float* __restrict__ R,
                                                   const float* __restrict__ W, const float* __restrict__ bias,
                                                   float* __restrict__ g, int D1, int D2, int rows) {
  int lane = threadIdx.x & 31;
  int row = blockIdx.x * 8 + (threadIdx.x >> 5);
  if (row >= rows) return;
  const float* x = X + (size_t)row * D1;
  const float* r = R + (size_t)row * D2;
  float s = 0.f;
  for (int i = lane; i < D1; i += 32) s = fmaf(x[i], W[i], s);
  for (int i = lane; i < D2; i += 32) s = fmaf(r[i], W[D1 + i], s);
  for (int o = 16; o > 0; o >>= 1) s += __shfl_xor(s, o, 32);
  if (lane == 0) g[row] = 1.0f / (1.0f + expf(-(s + bias[0])));
}

// ---------------- JAX threefry2x32 + normal (key 42) ----------------
__device__ __forceinline__ float erfinv_giles(float x) {
  float w = -logf(fmaf(-x, x, 1.0f));   // -log1p(-x^2) with fused 1-x^2
  float p;
  if (w < 5.0f) {
    w -= 2.5f;
    p = 2.81022636e-08f;
    p = fmaf(p, w, 3.43273939e-07f);
    p = fmaf(p, w, -3.5233877e-06f);
    p = fmaf(p, w, -4.39150654e-06f);
    p = fmaf(p, w, 0.00021858087f);
    p = fmaf(p, w, -0.00125372503f);
    p = fmaf(p, w, -0.00417768164f);
    p = fmaf(p, w, 0.246640727f);
    p = fmaf(p, w, 1.50140941f);
  } else {
    w = sqrtf(w) - 3.0f;
    p = -0.000200214257f;
    p = fmaf(p, w, 0.000100950558f);
    p = fmaf(p, w, 0.00134934322f);
    p = fmaf(p, w, -0.00367342844f);
    p = fmaf(p, w, 0.00573950773f);
    p = fmaf(p, w, -0.0076224613f);
    p = fmaf(p, w, 0.00943887047f);
    p = fmaf(p, w, 1.00167406f);
    p = fmaf(p, w, 2.83297682f);
  }
  return p * x;
}

__device__ __forceinline__ float bits_to_normal(unsigned bits) {
  float f = __uint_as_float((bits >> 9) | 0x3f800000u) - 1.0f;  // [0,1)
  const float lo = -0.99999994f;                                // nextafter(-1,0)
  float u = f * (1.0f - lo) + lo;
  u = fmaxf(u, lo);
  return 1.41421356237309505f * erfinv_giles(u);
}

#define TF_ROUND(a, b, r) { a += b; b = ((b) << (r)) | ((b) >> (32 - (r))); b ^= a; }

__global__ void eps_kernel(__hip_bfloat16* __restrict__ eps, unsigned n2, unsigned k0, unsigned k1) {
  unsigned i = blockIdx.x * blockDim.x + threadIdx.x;
  if (i >= n2) return;
  unsigned ks2 = k0 ^ k1 ^ 0x1BD11BDAu;
  unsigned x0 = i + k0, x1 = (i + n2) + k1;
  TF_ROUND(x0, x1, 13) TF_ROUND(x0, x1, 15) TF_ROUND(x0, x1, 26) TF_ROUND(x0, x1, 6)
  x0 += k1;  x1 += ks2 + 1u;
  TF_ROUND(x0, x1, 17) TF_ROUND(x0, x1, 29) TF_ROUND(x0, x1, 16) TF_ROUND(x0, x1, 24)
  x0 += ks2; x1 += k0 + 2u;
  TF_ROUND(x0, x1, 13) TF_ROUND(x0, x1, 15) TF_ROUND(x0, x1, 26) TF_ROUND(x0, x1, 6)
  x0 += k0;  x1 += k1 + 3u;
  TF_ROUND(x0, x1, 17) TF_ROUND(x0, x1, 29) TF_ROUND(x0, x1, 16) TF_ROUND(x0, x1, 24)
  x0 += k1;  x1 += ks2 + 4u;
  TF_ROUND(x0, x1, 13) TF_ROUND(x0, x1, 15) TF_ROUND(x0, x1, 26) TF_ROUND(x0, x1, 6)
  x0 += ks2; x1 += k0 + 5u;
  eps[i]      = __float2bfloat16(bits_to_normal(x0));
  eps[i + n2] = __float2bfloat16(bits_to_normal(x1));
}

// ---------------- host side ----------------
static inline void launch_gemm(const GemmP& p, int nB, hipStream_t s) {
  dim3 grid(p.M / 64, p.N / 64, nB);
  gemm_bf16_wmma<<<grid, dim3(256), 0, s>>>(p);
}

extern "C" void kernel_launch(void* const* d_in, const int* in_sizes, int n_in,
                              void* d_out, int out_size, void* d_ws, size_t ws_size,
                              hipStream_t stream) {
  (void)in_sizes; (void)n_in; (void)out_size; (void)ws_size;
  const int nB = 16, T = 1024, DA = 1024, DV = 512, DM = 1024, DF = 1536;
  const float EPS = 1e-6f;

  const float* Ain  = (const float*)d_in[0];
  const float* Vin  = (const float*)d_in[1];
  const float* Qaw  = (const float*)d_in[2];  const float* Qab  = (const float*)d_in[3];
  const float* Kvw  = (const float*)d_in[4];  const float* Kvb  = (const float*)d_in[5];
  const float* Vvw  = (const float*)d_in[6];  const float* Vvb  = (const float*)d_in[7];
  const float* Qvw  = (const float*)d_in[8];  const float* Qvb  = (const float*)d_in[9];
  const float* Kaw  = (const float*)d_in[10]; const float* Kab  = (const float*)d_in[11];
  const float* Vaw  = (const float*)d_in[12]; const float* Vab  = (const float*)d_in[13];
  const float* WAg  = (const float*)d_in[14]; const float* bAg  = (const float*)d_in[15];
  const float* WVg  = (const float*)d_in[16]; const float* bVg  = (const float*)d_in[17];
  const float* v2aw = (const float*)d_in[18]; const float* v2ab = (const float*)d_in[19];
  const float* a2vw = (const float*)d_in[20]; const float* a2vb = (const float*)d_in[21];
  float* Out = (float*)d_out;

  char* base = (char*)d_ws;
  size_t off = 0;
  auto alloc = [&](size_t bytes) -> char* {
    char* p = base + off;
    off = (off + bytes + 255) & ~(size_t)255;
    return p;
  };
  typedef __hip_bfloat16 bf;
  // bf16 buffers
  bf* Ab     = (bf*)alloc((size_t)nB * T * DA * 2);
  bf* Vb     = (bf*)alloc((size_t)nB * T * DV * 2);
  bf* wQa    = (bf*)alloc((size_t)DM * DA * 2);
  bf* wKv    = (bf*)alloc((size_t)DM * DV * 2);
  bf* wVv    = (bf*)alloc((size_t)DV * DV * 2);
  bf* wQv    = (bf*)alloc((size_t)DM * DV * 2);
  bf* wKa    = (bf*)alloc((size_t)DM * DA * 2);
  bf* wVa    = (bf*)alloc((size_t)DA * DA * 2);
  bf* wv2a   = (bf*)alloc((size_t)DA * DV * 2);
  bf* wa2v   = (bf*)alloc((size_t)DV * DA * 2);
  bf* Qa     = (bf*)alloc((size_t)nB * T * DM * 2);
  bf* Kv     = (bf*)alloc((size_t)nB * T * DM * 2);
  bf* VvT    = (bf*)alloc((size_t)nB * DV * T * 2);
  bf* Qv     = (bf*)alloc((size_t)nB * T * DM * 2);
  bf* Ka     = (bf*)alloc((size_t)nB * T * DM * 2);
  bf* VaT    = (bf*)alloc((size_t)nB * DA * T * 2);
  bf* Acb    = (bf*)alloc((size_t)nB * T * DA * 2);
  bf* AcT    = (bf*)alloc((size_t)nB * DA * T * 2);
  bf* Vcb    = (bf*)alloc((size_t)nB * T * DV * 2);
  bf* VcT    = (bf*)alloc((size_t)nB * DV * T * 2);
  bf* MinvAb = (bf*)alloc((size_t)nB * DA * DA * 2);   // later reused for bf16(L_a)
  bf* MinvVb = (bf*)alloc((size_t)nB * DV * DV * 2);   // later reused for bf16(L_v)
  bf* Pb     = (bf*)alloc((size_t)nB * T * T * 2);
  bf* resAvB = (bf*)alloc((size_t)nB * T * DV * 2);
  bf* resVaB = (bf*)alloc((size_t)nB * T * DA * 2);
  bf* epsB   = (bf*)alloc((size_t)nB * T * DF * 2);
  // f32 buffers
  float* muA    = (float*)alloc((size_t)nB * DA * 4);
  float* muV    = (float*)alloc((size_t)nB * DV * 4);
  float* SigA   = (float*)alloc((size_t)nB * DA * DA * 4);   // -> L_a in place
  float* SigV   = (float*)alloc((size_t)nB * DV * DV * 4);   // -> L_v in place
  float* MinvA  = (float*)alloc((size_t)nB * DA * DA * 4);   // reused as attention S (B*T*T == B*DA*DA)
  float* MinvV  = (float*)alloc((size_t)nB * DV * DV * 4);
  float* logdA  = (float*)alloc((size_t)nB * 4);
  float* logdV  = (float*)alloc((size_t)nB * 4);
  float* mahaA  = (float*)alloc((size_t)nB * T * 4);
  float* mahaV  = (float*)alloc((size_t)nB * T * 4);
  float* pdfA   = (float*)alloc((size_t)nB * T * 4);
  float* pdfV   = (float*)alloc((size_t)nB * T * 4);
  float* resAv  = (float*)alloc((size_t)nB * T * DV * 4);
  float* resVa  = (float*)alloc((size_t)nB * T * DA * 4);
  float* gA     = (float*)alloc((size_t)nB * T * 4);
  float* gV     = (float*)alloc((size_t)nB * T * 4);
  float* Sbuf   = MinvA;  // reuse: Minv f32 consumed (as bf16 copy) before S is produced

  auto f2b = [&](const float* src, bf* dst, size_t n) {
    int blocks = (int)((n + 2047) / 2048); if (blocks > 16384) blocks = 16384;
    f2b_kernel<<<blocks, 256, 0, stream>>>(src, dst, n);
  };
  auto zero = [&](float* dst, size_t n) {
    int blocks = (int)((n + 2047) / 2048); if (blocks > 16384) blocks = 16384;
    zero_kernel<<<blocks, 256, 0, stream>>>(dst, n);
  };

  // ---- stage 0: bf16 conversions of streaming operands and weights ----
  f2b(Ain, Ab, (size_t)nB * T * DA);
  f2b(Vin, Vb, (size_t)nB * T * DV);
  f2b(Qaw, wQa, (size_t)DM * DA);   f2b(Kvw, wKv, (size_t)DM * DV);
  f2b(Vvw, wVv, (size_t)DV * DV);   f2b(Qvw, wQv, (size_t)DM * DV);
  f2b(Kaw, wKa, (size_t)DM * DA);   f2b(Vaw, wVa, (size_t)DA * DA);
  f2b(v2aw, wv2a, (size_t)DA * DV); f2b(a2vw, wa2v, (size_t)DV * DA);

  // ---- stage 1: means, centered (+transposed) bf16 ----
  mean_kernel<<<dim3(DA / 256, nB), 256, 0, stream>>>(Ain, muA, T, DA);
  mean_kernel<<<dim3(DV / 256, nB), 256, 0, stream>>>(Vin, muV, T, DV);
  center_tb_kernel<<<dim3(DA / 32, T / 32, nB), dim3(32, 32), 0, stream>>>(Ain, muA, Acb, AcT, T, DA);
  center_tb_kernel<<<dim3(DV / 32, T / 32, nB), dim3(32, 32), 0, stream>>>(Vin, muV, Vcb, VcT, T, DV);

  // ---- stage 2: covariance GEMMs  Sigma = Xc^T Xc/(T-1) + 2*EPS*I ----
  {
    GemmP p = {}; p.A = (const __bf16*)AcT; p.sA = (size_t)DA * T; p.lda = T;
    p.Bm = (const __bf16*)AcT; p.sB = (size_t)DA * T; p.ldb = T;
    p.outF = SigA; p.sO = (size_t)DA * DA; p.ldo = DA;
    p.M = DA; p.N = DA; p.K = T; p.alpha = 1.0f / (T - 1); p.diagAdd = 2.0f * EPS; p.mode = EPI_F32;
    launch_gemm(p, nB, stream);
  }
  {
    GemmP p = {}; p.A = (const __bf16*)VcT; p.sA = (size_t)DV * T; p.lda = T;
    p.Bm = (const __bf16*)VcT; p.sB = (size_t)DV * T; p.ldb = T;
    p.outF = SigV; p.sO = (size_t)DV * DV; p.ldo = DV;
    p.M = DV; p.N = DV; p.K = T; p.alpha = 1.0f / (T - 1); p.diagAdd = 2.0f * EPS; p.mode = EPI_F32;
    launch_gemm(p, nB, stream);
  }

  // ---- stage 3: Cholesky (shared by pdf path and final sampling), logdet, L^-1 ----
  cholesky_kernel<<<nB, 256, 0, stream>>>(SigA, (size_t)DA * DA, DA);
  cholesky_kernel<<<nB, 256, 0, stream>>>(SigV, (size_t)DV * DV, DV);
  logdet_kernel<<<nB, 256, 0, stream>>>(SigA, logdA, DA);
  logdet_kernel<<<nB, 256, 0, stream>>>(SigV, logdV, DV);
  trinv_kernel<<<nB, 256, 0, stream>>>(SigA, MinvA, (size_t)DA * DA, DA);
  trinv_kernel<<<nB, 256, 0, stream>>>(SigV, MinvV, (size_t)DV * DV, DV);
  f2b(MinvA, MinvAb, (size_t)nB * DA * DA);
  f2b(MinvV, MinvVb, (size_t)nB * DV * DV);

  // ---- stage 4: Mahalanobis via GEMM epilogue (atomic row sum of squares) ----
  zero(mahaA, (size_t)nB * T);
  zero(mahaV, (size_t)nB * T);
  {
    GemmP p = {}; p.A = (const __bf16*)Acb; p.sA = (size_t)T * DA; p.lda = DA;
    p.Bm = (const __bf16*)MinvAb; p.sB = (size_t)DA * DA; p.ldb = DA;
    p.M = T; p.N = DA; p.K = DA; p.maha = mahaA; p.sMaha = T; p.mode = EPI_MAHA;
    launch_gemm(p, nB, stream);
  }
  {
    GemmP p = {}; p.A = (const __bf16*)Vcb; p.sA = (size_t)T * DV; p.lda = DV;
    p.Bm = (const __bf16*)MinvVb; p.sB = (size_t)DV * DV; p.ldb = DV;
    p.M = T; p.N = DV; p.K = DV; p.maha = mahaV; p.sMaha = T; p.mode = EPI_MAHA;
    launch_gemm(p, nB, stream);
  }
  pdf_kernel<<<(nB * T) / 256, 256, 0, stream>>>(mahaA, logdA, pdfA, DA, T, nB * T);
  pdf_kernel<<<(nB * T) / 256, 256, 0, stream>>>(mahaV, logdV, pdfV, DV, T, nB * T);

  // ---- stage 5: the six linear projections (bf16 WMMA, fused bias) ----
  auto proj = [&](const bf* X, int K, const bf* W, const float* bias, int N, bf* Yout, int mode, int ldo, size_t sO) {
    GemmP p = {}; p.A = (const __bf16*)X; p.sA = (size_t)T * K; p.lda = K;
    p.Bm = (const __bf16*)W; p.sB = 0; p.ldb = K; p.bias = bias;
    p.outB = Yout; p.sO = sO; p.ldo = ldo; p.M = T; p.N = N; p.K = K; p.mode = mode;
    launch_gemm(p, nB, stream);
  };
  proj(Ab, DA, wQa, Qab, DM, Qa,  EPI_BF16,   DM, (size_t)T * DM);
  proj(Vb, DV, wKv, Kvb, DM, Kv,  EPI_BF16,   DM, (size_t)T * DM);
  proj(Vb, DV, wVv, Vvb, DV, VvT, EPI_BF16_T, T,  (size_t)DV * T);  // stored (DV,T) for P@V
  proj(Vb, DV, wQv, Qvb, DM, Qv,  EPI_BF16,   DM, (size_t)T * DM);
  proj(Ab, DA, wKa, Kab, DM, Ka,  EPI_BF16,   DM, (size_t)T * DM);
  proj(Ab, DA, wVa, Vab, DA, VaT, EPI_BF16_T, T,  (size_t)DA * T);

  // ---- stage 6: PDF-weighted cross attention (two-pass, S reuses Minv f32 region) ----
  auto attention = [&](const bf* Q, const bf* K, const bf* VT, int Dout,
                       const float* keyw, float* resF, bf* resB) {
    GemmP s = {}; s.A = (const __bf16*)Q; s.sA = (size_t)T * DM; s.lda = DM;
    s.Bm = (const __bf16*)K; s.sB = (size_t)T * DM; s.ldb = DM;
    s.outF = Sbuf; s.sO = (size_t)T * T; s.ldo = T;
    s.M = T; s.N = T; s.K = DM; s.alpha = 0.03125f /* 1/sqrt(1024) */;
    s.colScale = keyw; s.sCol = T; s.mode = EPI_F32;
    launch_gemm(s, nB, stream);
    softmax_kernel<<<(nB * T) / 8, 256, 0, stream>>>(Sbuf, Pb, T, nB * T);
    GemmP o = {}; o.A = (const __bf16*)Pb; o.sA = (size_t)T * T; o.lda = T;
    o.Bm = (const __bf16*)VT; o.sB = (size_t)Dout * T; o.ldb = T;
    o.outF = resF; o.sO = (size_t)T * Dout; o.ldo = Dout;
    o.M = T; o.N = Dout; o.K = T; o.alpha = 1.0f; o.mode = EPI_F32;
    launch_gemm(o, nB, stream);
    f2b(resF, resB, (size_t)nB * T * Dout);
  };
  attention(Qa, Kv, VvT, DV, pdfV, resAv, resAvB);   // audio -> visual
  attention(Qv, Ka, VaT, DA, pdfA, resVa, resVaB);   // visual -> audio

  // ---- stage 7: gates and gated fusion written straight into d_out (= mu_f) ----
  gate_kernel<<<(nB * T) / 8, 256, 0, stream>>>(Ain, resAv, WAg, bAg, gA, DA, DV, nB * T);
  gate_kernel<<<(nB * T) / 8, 256, 0, stream>>>(Vin, resVa, WVg, bVg, gV, DV, DA, nB * T);
  {
    GemmP p = {}; p.A = (const __bf16*)resAvB; p.sA = (size_t)T * DV; p.lda = DV;
    p.Bm = (const __bf16*)wv2a; p.sB = 0; p.ldb = DV; p.bias = v2ab;
    p.outF = Out; p.sO = (size_t)T * DF; p.ldo = DF;
    p.gate = gA; p.sGate = T; p.skip = Ain; p.sSkip = (size_t)T * DA; p.ldskip = DA;
    p.M = T; p.N = DA; p.K = DV; p.mode = EPI_GATED;
    launch_gemm(p, nB, stream);
  }
  {
    GemmP p = {}; p.A = (const __bf16*)resVaB; p.sA = (size_t)T * DA; p.lda = DA;
    p.Bm = (const __bf16*)wa2v; p.sB = 0; p.ldb = DA; p.bias = a2vb;
    p.outF = Out + DA; p.sO = (size_t)T * DF; p.ldo = DF;
    p.gate = gV; p.sGate = T; p.skip = Vin; p.sSkip = (size_t)T * DV; p.ldskip = DV;
    p.M = T; p.N = DV; p.K = DA; p.mode = EPI_GATED;
    launch_gemm(p, nB, stream);
  }

  // ---- stage 8: reparameterized sampling out += eps @ L^T (block-diagonal L) ----
  f2b(SigA, MinvAb, (size_t)nB * DA * DA);   // bf16(L_a), reuse buffer
  f2b(SigV, MinvVb, (size_t)nB * DV * DV);   // bf16(L_v)
  {
    unsigned n = (unsigned)nB * T * DF, n2 = n / 2;
    eps_kernel<<<(n2 + 255) / 256, 256, 0, stream>>>(epsB, n2, 0u, 42u);  // jax.random.key(42)
  }
  {
    GemmP p = {}; p.A = (const __bf16*)epsB; p.sA = (size_t)T * DF; p.lda = DF;
    p.Bm = (const __bf16*)MinvAb; p.sB = (size_t)DA * DA; p.ldb = DA;
    p.outF = Out; p.sO = (size_t)T * DF; p.ldo = DF;
    p.M = T; p.N = DA; p.K = DA; p.mode = EPI_ACC;
    launch_gemm(p, nB, stream);
  }
  {
    GemmP p = {}; p.A = (const __bf16*)(epsB + DA); p.sA = (size_t)T * DF; p.lda = DF;
    p.Bm = (const __bf16*)MinvVb; p.sB = (size_t)DV * DV; p.ldb = DV;
    p.outF = Out + DA; p.sO = (size_t)T * DF; p.ldo = DF;
    p.M = T; p.N = DV; p.K = DV; p.mode = EPI_ACC;
    launch_gemm(p, nB, stream);
  }
}
